// MultiheadAttention_80702435491877
// MI455X (gfx1250) — matrix operand order repaired and checked by reading, not compiled, not run on hardware
//
#include <hip/hip_runtime.h>

// ---------------------------------------------------------------------------
// GPT-2 style MHA for MI455X (gfx1250): all GEMMs via v_wmma_f32_16x16x32_bf16
// + Tensor Data Mover (tensor_load_to_lds) for the bf16 A-tile in the
//   output-projection GEMM.
// ---------------------------------------------------------------------------

typedef __attribute__((ext_vector_type(16))) __bf16 v16bf;
typedef __attribute__((ext_vector_type(8)))  float  v8f;
typedef unsigned int uint32x4 __attribute__((ext_vector_type(4)));
typedef unsigned int uint32x8 __attribute__((ext_vector_type(8)));

#define S_LEN  4096
#define DMODEL 768
#define NH     12
#define HD     64
#define QKV_N  (3 * DMODEL)

// fp32 -> bf16, round-to-nearest-even
__device__ __forceinline__ unsigned short f2bf(float f) {
    unsigned int u = __float_as_uint(f);
    u += 0x7fffu + ((u >> 16) & 1u);
    return (unsigned short)(u >> 16);
}
__device__ __forceinline__ unsigned int pk2bf(float lo, float hi) {
    return (unsigned int)f2bf(lo) | ((unsigned int)f2bf(hi) << 16);
}

union BfFrag {
    v16bf        v;
    unsigned int u[8];
};

// ---------------------------------------------------------------------------
// TDM: DMA a [tile_d1 rows x tile_d0 elems] bf16 tile (row stride = stride0
// elements) from global memory into LDS at byte offset ldsOff.
// D# layout per CDNA5 ISA ch. 8 (groups 0 and 1; VADDR2/3 = NULL -> 2D).
// ---------------------------------------------------------------------------
__device__ __forceinline__ void tdm_load_tile_bf16(
    const unsigned short* gtile, unsigned ldsOff,
    unsigned tensor_d0, unsigned tensor_d1,
    unsigned tile_d0, unsigned tile_d1, unsigned stride0)
{
    unsigned long long ga = (unsigned long long)gtile;
    uint32x4 g0;
    g0[0] = 1u;                                   // count=1, is_restore=0, gather=0
    g0[1] = ldsOff;                               // lds_addr
    g0[2] = (unsigned)(ga & 0xffffffffu);         // global_addr[31:0]
    g0[3] = (unsigned)(ga >> 32) | 0x80000000u;   // global_addr[56:32] | type=2
    uint32x8 g1;
    g1[0] = 0x00010000u;                          // wg_mask=0, data_size=1 (2B)
    g1[1] = (tensor_d0 & 0xffffu) << 16;          // tensor_dim0[15:0]
    g1[2] = (tensor_d0 >> 16) | ((tensor_d1 & 0xffffu) << 16);
    g1[3] = (tensor_d1 >> 16) | (tile_d0 << 16);  // tile_dim0
    g1[4] = tile_d1;                              // tile_dim1 (tile_dim2=0)
    g1[5] = stride0;                              // tensor_dim0_stride[31:0]
    g1[6] = 0u;                                   // stride hi / dim1_stride lo
    g1[7] = 0u;
    asm volatile("tensor_load_to_lds %0, %1" :: "s"(g0), "s"(g1) : "memory");
}
__device__ __forceinline__ void tdm_wait() {
    asm volatile("s_wait_tensorcnt 0x0" ::: "memory");
}

// ---------------------------------------------------------------------------
// Kernel 1: qkv = X @ Wqkv + b ; scatter into Q[h][s][d], K[h][s][d], Vt[h][d][s]
//   Workgroup tile: 64(M) x 64(N), 4 waves, each wave = 16 rows x 64 cols.
// ---------------------------------------------------------------------------
__global__ __launch_bounds__(128)
void qkv_gemm_kernel(const float* __restrict__ X, const float* __restrict__ W,
                     const float* __restrict__ bias,
                     unsigned short* __restrict__ Qb,
                     unsigned short* __restrict__ Kb,
                     unsigned short* __restrict__ Vt)
{
    __shared__ unsigned short As[64 * 32];   // [m_local][k_local]
    __shared__ unsigned short Bs[64 * 32];   // [n_local][k_local]

    const int tid  = threadIdx.x;
    const int lane = tid & 31;
    const int wave = tid >> 5;
    const int g    = lane >> 4;     // half-wave group
    const int lr   = lane & 15;
    const int n0   = blockIdx.x * 64;
    const int m0   = blockIdx.y * 64;

    v8f acc[4];
#pragma unroll
    for (int t = 0; t < 4; ++t)
#pragma unroll
        for (int r = 0; r < 8; ++r) acc[t][r] = 0.0f;

    for (int k0 = 0; k0 < DMODEL; k0 += 32) {
        // stage A tile (fp32 -> bf16), float4 loads / packed b64 stores
#pragma unroll
        for (int j = 0; j < 4; ++j) {
            int idx = (tid + j * 128) * 4;
            int row = idx >> 5, col = idx & 31;
            const float4 f = *(const float4*)&X[(size_t)(m0 + row) * DMODEL + k0 + col];
            *(unsigned int*)&As[idx]     = pk2bf(f.x, f.y);
            *(unsigned int*)&As[idx + 2] = pk2bf(f.z, f.w);
        }
        // stage B tile transposed: Bs[n][k], float4 loads / b16 scatter
#pragma unroll
        for (int j = 0; j < 4; ++j) {
            int idx = (tid + j * 128) * 4;
            int kk = idx >> 6, nn = idx & 63;
            const float4 f = *(const float4*)&W[(size_t)(k0 + kk) * QKV_N + n0 + nn];
            Bs[(nn + 0) * 32 + kk] = f2bf(f.x);
            Bs[(nn + 1) * 32 + kk] = f2bf(f.y);
            Bs[(nn + 2) * 32 + kk] = f2bf(f.z);
            Bs[(nn + 3) * 32 + kk] = f2bf(f.w);
        }
        if (k0 + 32 < DMODEL) {
            __builtin_prefetch(&X[(size_t)(m0 + (tid >> 1)) * DMODEL + k0 + 32], 0, 1);
            __builtin_prefetch(&W[(size_t)(k0 + 32 + (tid >> 2)) * QKV_N + n0], 0, 1);
        }
        __syncthreads();

        // A fragment: 16x32, lane lr = row, element i -> k=g*16+i
        BfFrag a;
#pragma unroll
        for (int p = 0; p < 8; ++p) {
            int kk = g * 16 + 2 * p;
            a.u[p] = *(const unsigned int*)&As[(wave * 16 + lr) * 32 + kk];
        }
#pragma unroll
        for (int t = 0; t < 4; ++t) {
            // B fragment: 32x16, lane -> col n=lr, k = g*16 + i
            BfFrag b;
            const unsigned short* bp = &Bs[(t * 16 + lr) * 32 + g * 16];
#pragma unroll
            for (int p = 0; p < 8; ++p) b.u[p] = *(const unsigned int*)&bp[2 * p];
            acc[t] = __builtin_amdgcn_wmma_f32_16x16x32_bf16(
                false, a.v, false, b.v, (short)0, acc[t], false, false);
        }
        __syncthreads();
    }

    // epilogue: bias add, scatter to Q / K / V^T (bf16)
#pragma unroll
    for (int t = 0; t < 4; ++t) {
        int n   = n0 + t * 16 + lr;
        int seg = n / DMODEL;
        int nn  = n % DMODEL;
        int h   = nn / HD, d = nn % HD;
        float bv = bias[n];
#pragma unroll
        for (int r = 0; r < 8; ++r) {
            int m = m0 + wave * 16 + r + g * 8;
            unsigned short bfv = f2bf(acc[t][r] + bv);
            if (seg == 0)      Qb[((size_t)(h * S_LEN) + m) * HD + d] = bfv;
            else if (seg == 1) Kb[((size_t)(h * S_LEN) + m) * HD + d] = bfv;
            else               Vt[((size_t)(h * HD) + d) * S_LEN + m] = bfv;
        }
    }
}

// ---------------------------------------------------------------------------
// Kernel 2: flash attention. One wave handles (head h, 16 query rows).
//   Per 32-key step: 4 WMMA for S = Q K^T, online softmax, 4 WMMA for O += P V.
// ---------------------------------------------------------------------------
__global__ __launch_bounds__(32)
void attn_kernel(const unsigned short* __restrict__ Qb,
                 const unsigned short* __restrict__ Kb,
                 const unsigned short* __restrict__ Vt,
                 unsigned short* __restrict__ Aout)
{
    __shared__ unsigned short Ps[16 * 32];   // P tile staged for layout transpose

    const int lane  = threadIdx.x & 31;
    const int g     = lane >> 4;
    const int lr    = lane & 15;
    const int h     = blockIdx.y;
    const int qbase = blockIdx.x * 16;

    // Q A-fragments for d in [0,32) and [32,64)
    BfFrag qa[2];
    const unsigned short* qrow = Qb + ((size_t)(h * S_LEN) + qbase + lr) * HD;
#pragma unroll
    for (int c = 0; c < 2; ++c)
#pragma unroll
        for (int p = 0; p < 8; ++p) {
            int kk = c * 32 + g * 16 + 2 * p;
            qa[c].u[p] = *(const unsigned int*)&qrow[kk];
        }

    float mrun[8], lsum[8];
    v8f O[4];
#pragma unroll
    for (int r = 0; r < 8; ++r) { mrun[r] = -3.0e38f; lsum[r] = 0.0f; }
#pragma unroll
    for (int t = 0; t < 4; ++t)
#pragma unroll
        for (int r = 0; r < 8; ++r) O[t][r] = 0.0f;

    for (int kb = 0; kb < qbase + 16; kb += 32) {
        // ---- S = Q K^T : two 16x16 key tiles, contraction over d=64 ----
        v8f st[2];
#pragma unroll
        for (int t = 0; t < 2; ++t) {
#pragma unroll
            for (int r = 0; r < 8; ++r) st[t][r] = 0.0f;
#pragma unroll
            for (int c = 0; c < 2; ++c) {
                BfFrag kbf;   // B frag: n = key (lr), k = d = c*32 + g*16 + i
                const unsigned short* kr =
                    Kb + ((size_t)(h * S_LEN) + kb + t * 16 + lr) * HD + c * 32 + g * 16;
#pragma unroll
                for (int p = 0; p < 8; ++p) kbf.u[p] = *(const unsigned int*)&kr[2 * p];
                st[t] = __builtin_amdgcn_wmma_f32_16x16x32_bf16(
                    false, qa[c].v, false, kbf.v, (short)0, st[t], false, false);
            }
        }

        // ---- scale + causal mask + online softmax ----
        float pv[2][8];
#pragma unroll
        for (int r = 0; r < 8; ++r) {
            int q = qbase + r + g * 8;
            float s0 = st[0][r] * 0.125f;   // 1/sqrt(64)
            float s1 = st[1][r] * 0.125f;
            if (kb + lr > q)       s0 = -10000.0f;
            if (kb + 16 + lr > q)  s1 = -10000.0f;
            float mx = fmaxf(s0, s1);
#pragma unroll
            for (int off = 8; off >= 1; off >>= 1)
                mx = fmaxf(mx, __shfl_xor(mx, off, 32));
            float mnew  = fmaxf(mrun[r], mx);
            float alpha = __expf(mrun[r] - mnew);
            float p0 = __expf(s0 - mnew);
            float p1 = __expf(s1 - mnew);
            float rs = p0 + p1;
#pragma unroll
            for (int off = 8; off >= 1; off >>= 1)
                rs += __shfl_xor(rs, off, 32);
            lsum[r] = lsum[r] * alpha + rs;
            mrun[r] = mnew;
            pv[0][r] = p0; pv[1][r] = p1;
#pragma unroll
            for (int t = 0; t < 4; ++t) O[t][r] *= alpha;
        }

        // ---- transpose P (C-layout -> A-layout) through LDS ----
#pragma unroll
        for (int t = 0; t < 2; ++t)
#pragma unroll
            for (int r = 0; r < 8; ++r)
                Ps[(r + g * 8) * 32 + t * 16 + lr] = f2bf(pv[t][r]);
        __syncthreads();

        BfFrag pa;   // A frag: row = lr (query), k = key index in chunk
#pragma unroll
        for (int p = 0; p < 8; ++p) {
            int kk = g * 16 + 2 * p;
            pa.u[p] = *(const unsigned int*)&Ps[lr * 32 + kk];
        }

        // ---- O += P V : four 16-col tiles of V ----
#pragma unroll
        for (int t = 0; t < 4; ++t) {
            BfFrag vb;   // B frag: n = d col, k = key = kb + g*16 + i (contiguous in Vt)
            const unsigned short* vr =
                Vt + ((size_t)(h * HD) + t * 16 + lr) * S_LEN + kb + g * 16;
#pragma unroll
            for (int p = 0; p < 8; ++p) vb.u[p] = *(const unsigned int*)&vr[2 * p];
            O[t] = __builtin_amdgcn_wmma_f32_16x16x32_bf16(
                false, pa.v, false, vb.v, (short)0, O[t], false, false);
        }
        __syncthreads();
    }

    // ---- normalize and write merged-head output [s][h*64+d] (bf16) ----
#pragma unroll
    for (int t = 0; t < 4; ++t)
#pragma unroll
        for (int r = 0; r < 8; ++r) {
            int srow = qbase + r + g * 8;
            int col  = h * HD + t * 16 + lr;
            Aout[(size_t)srow * DMODEL + col] = f2bf(O[t][r] / lsum[r]);
        }
}

// ---------------------------------------------------------------------------
// Kernel 3: out = A @ Wproj + b  (A bf16 via TDM, W fp32->bf16, out fp32)
// ---------------------------------------------------------------------------
__global__ __launch_bounds__(128)
void proj_gemm_kernel(const unsigned short* __restrict__ A, const float* __restrict__ W,
                      const float* __restrict__ bias, float* __restrict__ out)
{
    __shared__ unsigned short As[64 * 32];
    __shared__ unsigned short Bs[64 * 32];

    const int tid  = threadIdx.x;
    const int lane = tid & 31;
    const int wave = tid >> 5;
    const int g    = lane >> 4;
    const int lr   = lane & 15;
    const int n0   = blockIdx.x * 64;
    const int m0   = blockIdx.y * 64;

    // LDS byte offset of As for the TDM descriptor
    const unsigned ldsAsOff =
        (unsigned)(unsigned long long)(__attribute__((address_space(3))) unsigned short*)As;

    v8f acc[4];
#pragma unroll
    for (int t = 0; t < 4; ++t)
#pragma unroll
        for (int r = 0; r < 8; ++r) acc[t][r] = 0.0f;

    for (int k0 = 0; k0 < DMODEL; k0 += 32) {
        // A tile: 64x32 bf16 copied by the Tensor Data Mover (wave 0 issues)
        if (tid < 32) {
            tdm_load_tile_bf16(A + (size_t)m0 * DMODEL + k0, ldsAsOff,
                               DMODEL, S_LEN, 32u, 64u, DMODEL);
        }
        // B tile: fp32 -> bf16 transpose staging by all threads
#pragma unroll
        for (int j = 0; j < 4; ++j) {
            int idx = (tid + j * 128) * 4;
            int kk = idx >> 6, nn = idx & 63;
            const float4 f = *(const float4*)&W[(size_t)(k0 + kk) * DMODEL + n0 + nn];
            Bs[(nn + 0) * 32 + kk] = f2bf(f.x);
            Bs[(nn + 1) * 32 + kk] = f2bf(f.y);
            Bs[(nn + 2) * 32 + kk] = f2bf(f.z);
            Bs[(nn + 3) * 32 + kk] = f2bf(f.w);
        }
        if (k0 + 32 < DMODEL) {
            __builtin_prefetch(&W[(size_t)(k0 + 32 + (tid >> 2)) * DMODEL + n0], 0, 1);
        }
        if (tid < 32) tdm_wait();
        __syncthreads();

        BfFrag a;
#pragma unroll
        for (int p = 0; p < 8; ++p) {
            int kk = g * 16 + 2 * p;
            a.u[p] = *(const unsigned int*)&As[(wave * 16 + lr) * 32 + kk];
        }
#pragma unroll
        for (int t = 0; t < 4; ++t) {
            BfFrag b;
            const unsigned short* bp = &Bs[(t * 16 + lr) * 32 + g * 16];
#pragma unroll
            for (int p = 0; p < 8; ++p) b.u[p] = *(const unsigned int*)&bp[2 * p];
            acc[t] = __builtin_amdgcn_wmma_f32_16x16x32_bf16(
                false, a.v, false, b.v, (short)0, acc[t], false, false);
        }
        __syncthreads();
    }

#pragma unroll
    for (int t = 0; t < 4; ++t) {
        int n = n0 + t * 16 + lr;
        float bv = bias[n];
#pragma unroll
        for (int r = 0; r < 8; ++r) {
            int m = m0 + wave * 16 + r + g * 8;
            out[(size_t)m * DMODEL + n] = acc[t][r] + bv;
        }
    }
}

// ---------------------------------------------------------------------------
extern "C" void kernel_launch(void* const* d_in, const int* in_sizes, int n_in,
                              void* d_out, int out_size, void* d_ws, size_t ws_size,
                              hipStream_t stream)
{
    (void)in_sizes; (void)n_in; (void)out_size; (void)ws_size;
    const float* X    = (const float*)d_in[0];   // [4096, 768]
    const float* Wqkv = (const float*)d_in[1];   // [768, 2304]
    const float* Bqkv = (const float*)d_in[2];   // [2304]
    const float* Wp   = (const float*)d_in[3];   // [768, 768]
    const float* Bp   = (const float*)d_in[4];   // [768]
    float* out = (float*)d_out;                  // [4096, 768]

    const size_t HEADSZ = (size_t)NH * S_LEN * HD;    // 3,145,728 bf16 elems
    unsigned short* ws   = (unsigned short*)d_ws;
    unsigned short* Qb   = ws;                 // Q[h][s][d]
    unsigned short* Kb   = ws + HEADSZ;        // K[h][s][d]
    unsigned short* Vt   = ws + 2 * HEADSZ;    // V^T[h][d][s]
    unsigned short* Aout = ws + 3 * HEADSZ;    // merged heads [s][768]

    qkv_gemm_kernel<<<dim3(QKV_N / 64, S_LEN / 64), 128, 0, stream>>>(
        X, Wqkv, Bqkv, Qb, Kb, Vt);
    attn_kernel<<<dim3(S_LEN / 16, NH), 32, 0, stream>>>(Qb, Kb, Vt, Aout);
    proj_gemm_kernel<<<dim3(DMODEL / 64, S_LEN / 64), 128, 0, stream>>>(
        Aout, Wp, Bp, out);
}
